// LSTMCell_3874060501273
// MI455X (gfx1250) — matrix operand order repaired and checked by reading, not compiled, not run on hardware
//
#include <hip/hip_runtime.h>

// Problem dimensions.
#define BDIM 8192
#define IDIM 2048
#define HDIM 2048
#define KCAT 4096            // concatenated K = I + H

// Tiling.
#define BM   256             // batch rows per block (2 strips of 16 per wave)
#define BN   32              // H columns per block (per gate)
#define KT   64              // K tile per LDS stage (2 WMMA K-steps)
#define ROWDW 34             // LDS dwords per tile row: 32 data + 2 pad (136B, 8B aligned)
#define ROWE  68             // bf16 elements per LDS row

// LDS layout (dword offsets): double-buffered A (256 rows) and W (128 rows) tiles.
#define A0_DW 0
#define A1_DW 8704           // 256 * 34
#define W0_DW 17408
#define W1_DW 21760          // + 128 * 34
#define SMEM_DW 26112        // 104,448 bytes (CDNA5: up to 320KB/WGP)

typedef __attribute__((ext_vector_type(16))) __bf16 v16bf;
typedef __attribute__((ext_vector_type(8)))  float  v8f;
typedef __attribute__((ext_vector_type(4)))  unsigned int u32x4;
typedef __attribute__((ext_vector_type(8)))  int i32x8;
typedef __attribute__((ext_vector_type(4)))  int i32x4;

#if defined(__has_builtin)
# if __has_builtin(__builtin_amdgcn_tensor_load_to_lds)
#  define HAVE_TDM 1
# endif
#endif
#ifndef HAVE_TDM
# define HAVE_TDM 0
#endif

union Frag16 {
    uint2 q[4];
    v16bf v;
};

__device__ __forceinline__ unsigned int bf16_rne(float f) {
    unsigned int u = __float_as_uint(f);
    unsigned int r = u + 0x7FFFu + ((u >> 16) & 1u);
    return r >> 16;
}
__device__ __forceinline__ unsigned int pack2_bf16(float lo, float hi) {
    return bf16_rne(lo) | (bf16_rne(hi) << 16);
}
__device__ __forceinline__ float sigmoid_f(float x) {
    return 1.0f / (1.0f + __expf(-x));
}
__device__ __forceinline__ float tanh_f(float x) {
    return 1.0f - 2.0f / (__expf(2.0f * x) + 1.0f);
}

// ---------------------------------------------------------------------------
// Kernel 1: fp32 -> bf16 pre-pack.  Abf[B,4096] = [x | h_prev],
// Wbf[4*H,4096] = [W | U].  Pure bandwidth (~384MB => ~16us @ 23.3TB/s).
// ---------------------------------------------------------------------------
__global__ __launch_bounds__(256)
void convert_pack(const float* __restrict__ x, const float* __restrict__ h_prev,
                  const float* __restrict__ W, const float* __restrict__ U,
                  unsigned short* __restrict__ Abf, unsigned short* __restrict__ Wbf)
{
    const unsigned chunk = blockIdx.x * 256u + threadIdx.x;  // 8 bf16 per thread
    const unsigned row = chunk >> 9;                         // 512 chunks per row
    const unsigned k   = (chunk & 511u) << 3;                // 0..4088
    const float* src;
    unsigned short* dst;
    if (row < BDIM) {
        src = (k < IDIM) ? (x + (size_t)row * IDIM + k)
                         : (h_prev + (size_t)row * HDIM + (k - IDIM));
        dst = Abf + (size_t)row * KCAT + k;
    } else {
        const unsigned r = row - BDIM;                       // g*2048 + h
        src = (k < IDIM) ? (W + (size_t)r * IDIM + k)
                         : (U + (size_t)r * HDIM + (k - IDIM));
        dst = Wbf + (size_t)r * KCAT + k;
    }
    float4 v0 = ((const float4*)src)[0];
    float4 v1 = ((const float4*)src)[1];
    uint4 o;
    o.x = pack2_bf16(v0.x, v0.y);
    o.y = pack2_bf16(v0.z, v0.w);
    o.z = pack2_bf16(v1.x, v1.y);
    o.w = pack2_bf16(v1.z, v1.w);
    *(uint4*)dst = o;
}

// ---------------------------------------------------------------------------
// TDM descriptor issue: tile [tile_d2][tile_d1][KT] bf16 from a row-major
// tensor with row stride KCAT elements, into padded LDS rows (32dw + 2dw pad).
// ---------------------------------------------------------------------------
__device__ __forceinline__ void tdm_issue(const unsigned short* gptr,
                                          unsigned ldsByte,
                                          unsigned tensor_d1, unsigned tile_d1,
                                          unsigned tile_d2,
                                          unsigned long long stride2)
{
#if HAVE_TDM
    unsigned long long ga = (unsigned long long)gptr;
    u32x4 g0;
    g0[0] = 1u;                                   // count=1, user descriptor
    g0[1] = ldsByte;                              // lds_addr
    g0[2] = (unsigned)(ga & 0xFFFFFFFFu);         // global_addr[31:0]
    g0[3] = (unsigned)((ga >> 32) & 0x01FFFFFFu) | (2u << 30);  // addr[56:32], type=2
    i32x8 g1;
    // data_size=2B, pad_enable, pad_interval=32dw(code 4), pad_amount=2dw(code 1)
    g1[0] = (1 << 16) | (1 << 20) | (4 << 22) | (1 << 25);
    g1[1] = (int)(((unsigned)KCAT & 0xFFFFu) << 16);               // tensor_dim0 lo
    g1[2] = (int)(((unsigned)KCAT >> 16) | ((tensor_d1 & 0xFFFFu) << 16));
    g1[3] = (int)((tensor_d1 >> 16) | ((unsigned)KT << 16));       // tile_dim0=KT
    g1[4] = (int)(tile_d1 | (tile_d2 << 16));                      // tile_dim1, tile_dim2
    g1[5] = (int)KCAT;                                             // dim0_stride lo
    g1[6] = 0;                                                     // dim0_stride hi | dim1_stride lo
    g1[7] = 0;
    i32x4 g2;
    g2[0] = (int)tile_d2;                                          // tensor_dim2 (0 => 2D)
    g2[1] = 0;
    g2[2] = (int)(stride2 & 0xFFFFFFFFull);                        // dim2_stride lo
    g2[3] = (int)((stride2 >> 32) & 0xFFFFull);                    // dim2_stride hi, tile_dim3=0
    i32x4 g3 = {0, 0, 0, 0};
    i32x8 g4 = {0, 0, 0, 0, 0, 0, 0, 0};   // 6-arg toolchain: zero-filled extra group
    __builtin_amdgcn_tensor_load_to_lds(g0, g1, g2, g3, g4, 0);
#else
    (void)gptr; (void)ldsByte; (void)tensor_d1; (void)tile_d1;
    (void)tile_d2; (void)stride2;
#endif
}

__device__ __forceinline__ Frag16 load_fa(const unsigned short* sA, int row, int ks,
                                          int hl) {
    // A fragment 16x32: lanes 0-15 M=row K={0..7,16..23}; lanes 16-31 K={8..15,24..31}
    Frag16 f;
    const unsigned short* p = sA + row * ROWE + ks * 32 + hl * 8;
    f.q[0] = *(const uint2*)(p + 0);
    f.q[1] = *(const uint2*)(p + 4);
    f.q[2] = *(const uint2*)(p + 16);
    f.q[3] = *(const uint2*)(p + 20);
    return f;
}
__device__ __forceinline__ Frag16 load_fb(const unsigned short* sW, int row, int ks,
                                          int hl) {
    // B fragment 32x16: lane=column, lanes 0-15 K={0..7,16..23}; lanes 16-31 K={8..15,24..31}
    Frag16 f;
    const unsigned short* p = sW + row * ROWE + ks * 32 + hl * 8;
    f.q[0] = *(const uint2*)(p + 0);
    f.q[1] = *(const uint2*)(p + 4);
    f.q[2] = *(const uint2*)(p + 16);
    f.q[3] = *(const uint2*)(p + 20);
    return f;
}

// ---------------------------------------------------------------------------
// Kernel 2: bf16 WMMA GEMM (both GEMMs fused via K=4096) + gate epilogue.
// Each wave owns two 16-row M strips so every B fragment feeds 2 WMMAs.
// ---------------------------------------------------------------------------
__global__ __launch_bounds__(256)
void lstm_gemm(const unsigned short* __restrict__ Abf,   // [B, 4096] bf16
               const unsigned short* __restrict__ Wbf,   // [4*H, 4096] bf16
               const float* __restrict__ C_prev,
               const float* __restrict__ bias,           // [4, H]
               float* __restrict__ out)                  // 5*B*H
{
    __shared__ unsigned int smem[SMEM_DW];

    const int tid  = threadIdx.x;
    const int lane = tid & 31;
    const int wave = tid >> 5;      // 0..7
    const int hl   = lane >> 4;
    const int l16  = lane & 15;

    const int n0 = blockIdx.x * BN; // H column base
    const int m0 = blockIdx.y * BM; // batch row base

    v8f acc[4][2][2];               // [gate][ntile][mstrip]
    const v8f vzero = {0.f, 0.f, 0.f, 0.f, 0.f, 0.f, 0.f, 0.f};
    #pragma unroll
    for (int g = 0; g < 4; ++g)
        #pragma unroll
        for (int t = 0; t < 2; ++t)
            #pragma unroll
            for (int ms = 0; ms < 2; ++ms)
                acc[g][t][ms] = vzero;

    const unsigned short* Arow0 = Abf + (size_t)m0 * KCAT;
    const unsigned short* Wrow0 = Wbf + (size_t)n0 * KCAT;   // gate g at +g*H*KCAT
    const unsigned long long strideW = (unsigned long long)HDIM * KCAT;
    const unsigned ldsBase = (unsigned)(size_t)(void*)&smem[0];

    constexpr int NSTEP = KCAT / KT;   // 64

#if HAVE_TDM
    if (wave == 0) {
        tdm_issue(Arow0, ldsBase + A0_DW * 4, (unsigned)(BDIM - m0), BM, 0, 0);
        tdm_issue(Wrow0, ldsBase + W0_DW * 4, (unsigned)(HDIM - n0), BN, 4, strideW);
    }
#endif

    #pragma unroll 1
    for (int step = 0; step < NSTEP; ++step) {
        const int buf = step & 1;
        const unsigned aDw = buf ? A1_DW : A0_DW;
        const unsigned wDw = buf ? W1_DW : W0_DW;

#if HAVE_TDM
        if (wave == 0) {
            if (step + 1 < NSTEP) {
                // prefetch next tile into the other buffer, then wait for this one
                tdm_issue(Arow0 + (size_t)(step + 1) * KT,
                          ldsBase + (buf ? A0_DW : A1_DW) * 4,
                          (unsigned)(BDIM - m0), BM, 0, 0);
                tdm_issue(Wrow0 + (size_t)(step + 1) * KT,
                          ldsBase + (buf ? W0_DW : W1_DW) * 4,
                          (unsigned)(HDIM - n0), BN, 4, strideW);
                __builtin_amdgcn_s_wait_tensorcnt(2);
            } else {
                __builtin_amdgcn_s_wait_tensorcnt(0);
            }
        }
        __syncthreads();
        asm volatile("" ::: "memory");   // TDM wrote LDS behind the compiler's back
#else
        // Fallback: cooperative bf16 global->LDS copy.
        __syncthreads();
        {
            const unsigned short* sa = Arow0 + (size_t)tid * KCAT + (size_t)step * KT;
            unsigned int* da = smem + aDw + tid * ROWDW;
            #pragma unroll
            for (int j = 0; j < 4; ++j) {
                uint4 va = ((const uint4*)sa)[j];
                da[j * 4 + 0] = va.x; da[j * 4 + 1] = va.y;
                da[j * 4 + 2] = va.z; da[j * 4 + 3] = va.w;
            }
            if (tid < 128) {
                const int g  = tid >> 5;
                const int wr = tid & 31;
                const unsigned short* sw =
                    Wrow0 + ((size_t)g * HDIM + wr) * KCAT + (size_t)step * KT;
                unsigned int* dw = smem + wDw + tid * ROWDW;
                #pragma unroll
                for (int j = 0; j < 4; ++j) {
                    uint4 vw = ((const uint4*)sw)[j];
                    dw[j * 4 + 0] = vw.x; dw[j * 4 + 1] = vw.y;
                    dw[j * 4 + 2] = vw.z; dw[j * 4 + 3] = vw.w;
                }
            }
        }
        __syncthreads();
#endif

        const unsigned short* sA = (const unsigned short*)smem + (size_t)aDw * 2;
        const unsigned short* sW = (const unsigned short*)smem + (size_t)wDw * 2;

        #pragma unroll
        for (int ks = 0; ks < 2; ++ks) {
            Frag16 fa0 = load_fa(sA, wave * 16 + l16, ks, hl);
            Frag16 fa1 = load_fa(sA, 128 + wave * 16 + l16, ks, hl);
            // Software-pipelined B fragments: load gt+1 while computing gt.
            Frag16 fb0 = load_fb(sW, 0 * 16 + l16, ks, hl);   // gt=0 (g=0,t=0)
            Frag16 fb1;
            #pragma unroll
            for (int gt = 0; gt < 8; ++gt) {
                const int g = gt >> 1;
                const int t = gt & 1;
                Frag16& cur = (gt & 1) ? fb1 : fb0;
                Frag16& nxt = (gt & 1) ? fb0 : fb1;
                if (gt < 7) {
                    const int g2 = (gt + 1) >> 1;
                    const int t2 = (gt + 1) & 1;
                    nxt = load_fb(sW, (g2 * BN) + t2 * 16 + l16, ks, hl);
                }
                acc[g][t][0] = __builtin_amdgcn_wmma_f32_16x16x32_bf16(
                    false, fa0.v, false, cur.v, (short)0, acc[g][t][0], false, false);
                acc[g][t][1] = __builtin_amdgcn_wmma_f32_16x16x32_bf16(
                    false, fa1.v, false, cur.v, (short)0, acc[g][t][1], false, false);
            }
        }
        __syncthreads();   // protect the buffer the next iteration overwrites
    }

    // ---- fused gate epilogue ----
    const size_t BH = (size_t)BDIM * HDIM;
    float* __restrict__ out_h  = out;
    float* __restrict__ out_C  = out + BH;
    float* __restrict__ out_i  = out + 2 * BH;
    float* __restrict__ out_f  = out + 3 * BH;
    float* __restrict__ out_ct = out + 4 * BH;

    #pragma unroll
    for (int t = 0; t < 2; ++t) {
        const int h = n0 + t * 16 + l16;
        const float bf = bias[0 * HDIM + h];
        const float bi = bias[1 * HDIM + h];
        const float bc = bias[2 * HDIM + h];
        const float bo = bias[3 * HDIM + h];
        #pragma unroll
        for (int ms = 0; ms < 2; ++ms) {
            #pragma unroll
            for (int r = 0; r < 8; ++r) {
                const int brow = m0 + ms * 128 + wave * 16 + hl * 8 + r;
                const size_t idx = (size_t)brow * HDIM + h;

                const float f_t = sigmoid_f(acc[0][t][ms][r] + bf);
                const float i_t = sigmoid_f(acc[1][t][ms][r] + bi);
                const float ctl = tanh_f   (acc[2][t][ms][r] + bc);
                const float o_t = sigmoid_f(acc[3][t][ms][r] + bo);

                const float Cp = C_prev[idx];
                const float Cn = f_t * Cp + i_t * ctl;
                const float hn = o_t * tanh_f(Cn);

                out_h [idx] = hn;
                out_C [idx] = Cn;
                out_i [idx] = i_t;
                out_f [idx] = f_t;
                out_ct[idx] = ctl;
            }
        }
    }
}

extern "C" void kernel_launch(void* const* d_in, const int* in_sizes, int n_in,
                              void* d_out, int out_size, void* d_ws, size_t ws_size,
                              hipStream_t stream) {
    (void)in_sizes; (void)n_in; (void)out_size; (void)ws_size;
    const float* x      = (const float*)d_in[0];
    const float* h_prev = (const float*)d_in[1];
    const float* C_prev = (const float*)d_in[2];
    const float* W      = (const float*)d_in[3];
    const float* U      = (const float*)d_in[4];
    const float* b      = (const float*)d_in[5];

    // Workspace: Abf (8192x4096 bf16, 64MB) + Wbf (8192x4096 bf16, 64MB) = 128MB.
    unsigned short* Abf = (unsigned short*)d_ws;
    unsigned short* Wbf = Abf + (size_t)BDIM * KCAT;

    // Pass 1: fp32 -> bf16 pack (fuses x|h_prev and W|U into K=4096 operands).
    const unsigned nchunks = (unsigned)(((size_t)(BDIM + 4 * HDIM) * KCAT) / 8);
    hipLaunchKernelGGL(convert_pack, dim3(nchunks / 256), dim3(256), 0, stream,
                       x, h_prev, W, U, Abf, Wbf);

    // Pass 2: TDM-fed WMMA GEMM + fused LSTM gate epilogue.
    hipLaunchKernelGGL(lstm_gemm, dim3(HDIM / BN, BDIM / BM), dim3(256), 0, stream,
                       Abf, Wbf, C_prev, b, (float*)d_out);
}
